// gRNN_17239998726650
// MI455X (gfx1250) — matrix operand order repaired and checked by reading, not compiled, not run on hardware
//
#include <hip/hip_runtime.h>
#include <hip/hip_bf16.h>

// Tree-LSTM for MI455X (gfx1250, wave32).
// - All U-matrix GEMMs: v_wmma_f32_16x16x32_bf16, f32 accumulate.
// - B operands staged in LDS via async global->LDS copies
//   (global_load_async_to_lds_b128 / s_wait_asynccnt), double-buffered so the
//   copy for n-block nb+1 overlaps the WMMAs of n-block nb.
// - Gate epilogue fused via LDS.
// Workspace requirement: ~302 MB (cA + cB + hA + hB + U_bf16), laid out below.

#define HSIZE   256
#define BATCH   64
#define NTIMES  4095      // 2^(11+1) - 1
#define DEPTH   11

// LDS strip geometry: 4 matrices x 16 rows x 256 K, rows padded to 264 elems
// (528 B = 132 words -> conflict-free ds_load_b128: half-wave hits banks 4n..4n+3)
#define BROW    264
#define BMAT    (16 * BROW)          // 4224 elems per matrix strip
#define BSTRIP  (4 * BMAT)           // 16896 elems per buffer

typedef __attribute__((ext_vector_type(16))) __bf16 v16bf;
typedef __attribute__((ext_vector_type(8)))  __bf16 v8bf;
typedef __attribute__((ext_vector_type(8)))  float  v8f;

__device__ __forceinline__ float sig_(float x) { return 1.0f / (1.0f + __expf(-x)); }
__device__ __forceinline__ float relu_(float x) { return fmaxf(x, 0.0f); }

// ---------------------------------------------------------------------------
// Convert the four 256x256 U matrices (f32, row-major (o,s)) to bf16, concat.
// ---------------------------------------------------------------------------
__global__ __launch_bounds__(256) void cvtU_kernel(
    const float* __restrict__ Uf, const float* __restrict__ Ui,
    const float* __restrict__ Uu, const float* __restrict__ Uo,
    __bf16* __restrict__ out)
{
    int i = blockIdx.x * 256 + threadIdx.x;        // 0 .. 65535
    out[i]          = (__bf16)Uf[i];
    out[i +  65536] = (__bf16)Ui[i];
    out[i + 131072] = (__bf16)Uu[i];
    out[i + 196608] = (__bf16)Uo[i];
}

// ---------------------------------------------------------------------------
// Leaf level: purely elementwise. Writes c (f32) and h (bf16) for 2048 leaves.
// ---------------------------------------------------------------------------
__global__ __launch_bounds__(256) void leaf_kernel(
    const float* __restrict__ times,
    const float* __restrict__ Wi_w, const float* __restrict__ Wi_b,
    const float* __restrict__ Wu_w, const float* __restrict__ Wu_b,
    const float* __restrict__ Wo_w, const float* __restrict__ Wo_b,
    __bf16* __restrict__ h_out, float* __restrict__ c_out)
{
    int idx  = blockIdx.x * 256 + threadIdx.x;     // b*2048*256 + node*256 + o
    int o    = idx & 255;
    int node = (idx >> 8) & 2047;
    int b    = idx >> 19;
    float t  = times[b * NTIMES + 2047 + node];    // times[:, (1<<DEPTH)-1 :]
    float iv = sig_(t * Wi_w[o] + Wi_b[o]);
    float uv = relu_(t * Wu_w[o] + Wu_b[o]);
    float c  = iv * uv;
    float ov = sig_(t * Wo_w[o] + Wo_b[o]);
    float h  = ov * relu_(c);
    c_out[idx] = c;
    h_out[idx] = (__bf16)h;
}

// ---------------------------------------------------------------------------
// One internal tree level. M = number of OUTPUT nodes (children = 2M).
// Block = 256 threads = 8 waves; block owns 16 (b,m) rows.
// Wave w: child = w>>2 (h[2m+child]), matrix = w&3 (Uf,Ui,Uu,Uo).
// Per n-block nb:
//   - strip (4 mats x 16 rows x 256 K, bf16) async-copied global->LDS,
//     double-buffered; copy for nb+1 issued before the WMMAs of nb
//   - 8 x v_wmma_f32_16x16x32_bf16 fed from LDS (ds_load_b128)
//   - D-tile -> LDS, barrier, fused gate epilogue, barrier
// ---------------------------------------------------------------------------
__global__ __launch_bounds__(256) void tree_level_kernel(
    const float* __restrict__ times,
    const float* __restrict__ Wi_w, const float* __restrict__ Wi_b,
    const float* __restrict__ Wu_w, const float* __restrict__ Wu_b,
    const float* __restrict__ Wo_w, const float* __restrict__ Wo_b,
    const float* __restrict__ Uf_b, const float* __restrict__ Ui_b,
    const float* __restrict__ Uu_b, const float* __restrict__ Uo_b,
    const __bf16* __restrict__ Umats,   // [4][256][256] bf16, order f,i,u,o
    const __bf16* __restrict__ h_in,    // [B][2M][256] bf16
    const float*  __restrict__ c_in,    // [B][2M][256] f32
    __bf16* __restrict__ h_out,         // [B][M][256] bf16
    float*  __restrict__ c_out,         // [B][M][256] f32
    int M)
{
    __shared__ __bf16 ldsB[2][BSTRIP];  // double-buffered B strips (~66 KB)
    __shared__ float  ldsY[8 * 256];    // 8 waves x (16x16) f32 D-tiles
    __shared__ float  ldsV[10 * 256];   // preloaded weight vectors, o-indexed

    const int tid  = threadIdx.x;
    const int wave = tid >> 5;
    const int lane = tid & 31;

    // Preload per-channel vectors (indexed by o = tid).
    ldsV[0 * 256 + tid] = Wi_w[tid];
    ldsV[1 * 256 + tid] = Wi_b[tid];
    ldsV[2 * 256 + tid] = Wu_w[tid];
    ldsV[3 * 256 + tid] = Wu_b[tid];
    ldsV[4 * 256 + tid] = Wo_w[tid];
    ldsV[5 * 256 + tid] = Wo_b[tid];
    ldsV[6 * 256 + tid] = Uf_b[tid];
    ldsV[7 * 256 + tid] = Ui_b[tid];
    ldsV[8 * 256 + tid] = Uu_b[tid];
    ldsV[9 * 256 + tid] = Uo_b[tid];

    const int r0 = blockIdx.x * 16;     // 16 rows of the (B*M) pair-row space

    // --- epilogue coordinates for this thread ---
    const int erow = tid >> 4;                  // 0..15
    const int ecol = tid & 15;                  // 0..15
    const int er   = r0 + erow;
    const int eb   = er / M;
    const int em   = er % M;
    const float t_e = times[eb * NTIMES + (M - 1) + em];   // times[:, M-1 : 2M-1]

    // --- async-copy chunk descriptors: 2048 x 16B chunks per strip, 8/thread.
    // Unpadded image: mat*8192 + row*512 + rowoff (bytes). LDS image padded
    // to BROW elems per row. Global row base shifts by nb*8192 bytes.
    unsigned ldsOff[8];
    unsigned gOff[8];
    #pragma unroll
    for (int j = 0; j < 8; ++j) {
        const unsigned boff   = (unsigned)(j * 256 + tid) * 16u;
        const unsigned cmat   = boff >> 13;
        const unsigned rem    = boff & 8191u;
        const unsigned crow   = rem >> 9;
        const unsigned rowoff = rem & 511u;
        gOff[j]   = cmat * 131072u + crow * 512u + rowoff;          // bytes
        ldsOff[j] = cmat * (BMAT * 2u) + crow * (BROW * 2u) + rowoff; // bytes
    }
    const unsigned long long ubase = (unsigned long long)(uintptr_t)Umats;
    const unsigned lbase0 = (unsigned)(uintptr_t)&ldsB[0][0];
    const unsigned lbase1 = (unsigned)(uintptr_t)&ldsB[1][0];

    // --- WMMA A-operand: 16 rows x 256 K of child h, bf16, ISA lane layout ---
    const int ch   = wave >> 2;                 // 0: h1 (even child), 1: h2 (odd)
    const int mat  = wave & 3;                  // 0..3 -> Uf,Ui,Uu,Uo
    const int arow = lane & 15;
    const int half = lane >> 4;
    const int ar   = r0 + arow;
    const int ab   = ar / M;
    const int am   = ar % M;
    const __bf16* aptr = h_in + (((size_t)ab * (2 * M) + (2 * am + ch)) << 8);

    v16bf A[8];
    #pragma unroll
    for (int ks = 0; ks < 8; ++ks) {
        const int k0 = ks * 32;
        v8bf lo = *(const v8bf*)(aptr + k0 + half * 8);
        v8bf hi = *(const v8bf*)(aptr + k0 + 16 + half * 8);
        #pragma unroll
        for (int e = 0; e < 8; ++e) { A[ks][e] = lo[e]; A[ks][8 + e] = hi[e]; }
    }

    // B-operand coordinates within the LDS strip (lane layout per ISA):
    // lanes 0-15: col n = lane, K k0..k0+7,k0+16..+23; lanes 16-31: n = lane-16, K k0+8..+15,k0+24..+31
    const int bn   = lane & 15;
    const int bkh  = half * 8;
    const int bele = mat * BMAT + bn * BROW;    // element base within strip

    // Kick off the async copy for nb = 0 into buffer 0.
    #pragma unroll
    for (int j = 0; j < 8; ++j) {
        unsigned long long ga = ubase + (unsigned long long)gOff[j];
        unsigned la = lbase0 + ldsOff[j];
        asm volatile("global_load_async_to_lds_b128 %0, %1, off"
                     :: "v"(la), "v"(ga) : "memory");
    }

    for (int nb = 0; nb < 16; ++nb) {
        const int cur = nb & 1;
        // Our portion of the current strip copy is done; barrier makes every
        // thread's portion visible before any ds_load of the strip.
        asm volatile("s_wait_asynccnt 0x0" ::: "memory");
        __syncthreads();

        // Prefetch next n-block's strip into the other buffer (overlaps WMMAs).
        if (nb < 15) {
            const unsigned gnb = (unsigned)(nb + 1) * 8192u;
            const unsigned lb  = cur ? lbase0 : lbase1;
            #pragma unroll
            for (int j = 0; j < 8; ++j) {
                unsigned long long ga = ubase + (unsigned long long)(gOff[j] + gnb);
                unsigned la = lb + ldsOff[j];
                asm volatile("global_load_async_to_lds_b128 %0, %1, off"
                             :: "v"(la), "v"(ga) : "memory");
            }
        }

        // ---- 16x16 output tile via 8 WMMAs fed from LDS ----
        v8f acc = {};
        #pragma unroll
        for (int ks = 0; ks < 8; ++ks) {
            const __bf16* bp = &ldsB[cur][bele + ks * 32 + bkh];
            v8bf b0 = *(const v8bf*)bp;
            v8bf b1 = *(const v8bf*)(bp + 16);
            v16bf Bm;
            #pragma unroll
            for (int e = 0; e < 8; ++e) { Bm[e] = b0[e]; Bm[8 + e] = b1[e]; }
            acc = __builtin_amdgcn_wmma_f32_16x16x32_bf16(
                      false, A[ks], false, Bm, (short)0, acc, false, false);
        }
        // Store D-tile: VGPR j, lane -> row = j + 8*(lane>>4), col = lane&15
        {
            float* dst = ldsY + wave * 256 + (half * 8) * 16 + bn;
            #pragma unroll
            for (int j = 0; j < 8; ++j) dst[j * 16] = acc[j];
        }
        __syncthreads();

        // ---- fused gate epilogue for columns o = nb*16 + ecol ----
        {
            const int o    = nb * 16 + ecol;
            const int yidx = erow * 16 + ecol;
            const float gf1 = ldsY[0 * 256 + yidx];
            const float gi1 = ldsY[1 * 256 + yidx];
            const float gu1 = ldsY[2 * 256 + yidx];
            const float go1 = ldsY[3 * 256 + yidx];
            const float gf2 = ldsY[4 * 256 + yidx];
            const float gi2 = ldsY[5 * 256 + yidx];
            const float gu2 = ldsY[6 * 256 + yidx];
            const float go2 = ldsY[7 * 256 + yidx];

            const float wiw = ldsV[0 * 256 + o], wib = ldsV[1 * 256 + o];
            const float wuw = ldsV[2 * 256 + o], wub = ldsV[3 * 256 + o];
            const float wow = ldsV[4 * 256 + o], wob = ldsV[5 * 256 + o];
            const float ufb = ldsV[6 * 256 + o];
            const float uib = ldsV[7 * 256 + o];
            const float uub = ldsV[8 * 256 + o];
            const float uob = ldsV[9 * 256 + o];

            const size_t cbase = ((size_t)eb * (2 * M) + 2 * em) << 8;
            const float c1 = c_in[cbase + o];
            const float c2 = c_in[cbase + 256 + o];

            const float f1 = sig_(gf1 + ufb);
            const float f2 = sig_(gf2 + ufb);
            // lin_t adds W*_b once; each of the two lin_h calls adds U*_b
            const float iv = sig_(t_e * wiw + wib + gi1 + gi2 + 2.0f * uib);
            const float uv = relu_(t_e * wuw + wub + gu1 + gu2 + 2.0f * uub);
            const float cn = iv * uv + f1 * c1 + f2 * c2;
            const float ov = sig_(t_e * wow + wob + go1 + go2 + 2.0f * uob);
            const float hn = ov * relu_(cn);

            const size_t oidx = (((size_t)eb * M + em) << 8) + o;
            c_out[oidx] = cn;
            h_out[oidx] = (__bf16)hn;
        }
        __syncthreads();   // ldsY reuse + strip-buffer reads all complete
    }
}

// ---------------------------------------------------------------------------
// Final projection: out[b,p] = h0[b,0,:] . P_w[p,:] + P_b[p]   (64 x 5)
// ---------------------------------------------------------------------------
__global__ void final_kernel(const __bf16* __restrict__ h0,
                             const float* __restrict__ P_w,
                             const float* __restrict__ P_b,
                             float* __restrict__ out)
{
    const int p = blockIdx.x;    // 0..4
    const int b = threadIdx.x;   // 0..63
    float s = P_b[p];
    #pragma unroll 8
    for (int k = 0; k < HSIZE; ++k)
        s += (float)h0[b * HSIZE + k] * P_w[p * HSIZE + k];
    out[b * 5 + p] = s;
}

// ---------------------------------------------------------------------------
extern "C" void kernel_launch(void* const* d_in, const int* in_sizes, int n_in,
                              void* d_out, int out_size, void* d_ws, size_t ws_size,
                              hipStream_t stream)
{
    (void)in_sizes; (void)n_in; (void)out_size; (void)ws_size;

    const float* times = (const float*)d_in[0];
    const float* Wi_w  = (const float*)d_in[1];
    const float* Wi_b  = (const float*)d_in[2];
    const float* Wu_w  = (const float*)d_in[3];
    const float* Wu_b  = (const float*)d_in[4];
    const float* Wo_w  = (const float*)d_in[5];
    const float* Wo_b  = (const float*)d_in[6];
    const float* Uf_w  = (const float*)d_in[7];
    const float* Uf_b  = (const float*)d_in[8];
    const float* Ui_w  = (const float*)d_in[9];
    const float* Ui_b  = (const float*)d_in[10];
    const float* Uu_w  = (const float*)d_in[11];
    const float* Uu_b  = (const float*)d_in[12];
    const float* Uo_w  = (const float*)d_in[13];
    const float* Uo_b  = (const float*)d_in[14];
    const float* P_w   = (const float*)d_in[15];
    const float* P_b   = (const float*)d_in[16];

    // Workspace layout (ping-pong c/h + bf16 U). Total = 302,514,176 bytes.
    char* ws = (char*)d_ws;
    const size_t szCA = (size_t)BATCH * 2048 * HSIZE * 4;   // 134,217,728
    const size_t szCB = (size_t)BATCH * 1024 * HSIZE * 4;   //  67,108,864
    const size_t szHA = (size_t)BATCH * 2048 * HSIZE * 2;   //  67,108,864
    const size_t szHB = (size_t)BATCH * 1024 * HSIZE * 2;   //  33,554,432
    float*  cA  = (float*) (ws);
    float*  cB  = (float*) (ws + szCA);
    __bf16* hA  = (__bf16*)(ws + szCA + szCB);
    __bf16* hB  = (__bf16*)(ws + szCA + szCB + szHA);
    __bf16* Ubf = (__bf16*)(ws + szCA + szCB + szHA + szHB);

    // 1) U matrices -> bf16 (tiny)
    cvtU_kernel<<<256, 256, 0, stream>>>(Uf_w, Ui_w, Uu_w, Uo_w, Ubf);

    // 2) Leaves -> (hA, cA) with 2048 nodes
    leaf_kernel<<<(BATCH * 2048 * HSIZE) / 256, 256, 0, stream>>>(
        times, Wi_w, Wi_b, Wu_w, Wu_b, Wo_w, Wo_b, hA, cA);

    // 3) Levels d = 10..0. d even (incl. 10): in = A, out = B; d odd: in = B, out = A.
    for (int d = DEPTH - 1; d >= 0; --d) {
        const int M = 1 << d;
        const __bf16* hin = (d & 1) ? hB : hA;
        const float*  cin = (d & 1) ? cB : cA;
        __bf16* hout = (d & 1) ? hA : hB;
        float*  cout = (d & 1) ? cA : cB;
        tree_level_kernel<<<4 * M, 256, 0, stream>>>(
            times, Wi_w, Wi_b, Wu_w, Wu_b, Wo_w, Wo_b,
            Uf_b, Ui_b, Uu_b, Uo_b, Ubf, hin, cin, hout, cout, M);
    }

    // 4) Final projection from level-0 h (in B buffer, M=1)
    final_kernel<<<5, 64, 0, stream>>>(hB, P_w, P_b, (float*)d_out);
}
